// MockModularAttention_10934986735922
// MI455X (gfx1250) — matrix operand order repaired and checked by reading, not compiled, not run on hardware
//
#include <hip/hip_runtime.h>

typedef __attribute__((ext_vector_type(16))) __bf16        v16bf;
typedef __attribute__((ext_vector_type(8)))  float         v8f;
typedef __attribute__((ext_vector_type(8)))  unsigned int  v8u;
typedef __attribute__((ext_vector_type(4)))  unsigned int  v4u;
typedef __attribute__((ext_vector_type(4)))  int           v4i;
typedef __attribute__((ext_vector_type(8)))  int           v8i;

#define BB 2
#define LL 2048
#define EE 512
#define HH 8
#define HD 64
#define MM (BB*LL)   /* 4096 rows of (B,L) */

// ---------- helpers ----------
static __device__ __forceinline__ unsigned short f2bf(float f) {
  unsigned u = __builtin_bit_cast(unsigned, f);
  return (unsigned short)((u + 0x7FFFu + ((u >> 16) & 1u)) >> 16);  // RNE
}
static __device__ __forceinline__ unsigned f2bf_pk(float lo, float hi) {
  return (unsigned)f2bf(lo) | ((unsigned)f2bf(hi) << 16);
}
static __device__ __forceinline__ v8f vzero() {
  v8f v;
#pragma unroll
  for (int i = 0; i < 8; i++) v[i] = 0.0f;
  return v;
}
static __device__ __forceinline__ v16bf asbf(v8u u) {
  return __builtin_bit_cast(v16bf, u);
}
static __device__ __forceinline__ v8f wmma_bf16(v16bf a, v16bf b, v8f c) {
  // D = A(16x32 bf16) x B(32x16 bf16) + C(16x16 f32)
  return __builtin_amdgcn_wmma_f32_16x16x32_bf16(false, a, false, b,
                                                 (short)0, c, false, false);
}

// ---------- TDM: async 2-D tile DMA global -> LDS (bf16 elements) ----------
// Builds a D# per CDNA5 ISA 8.3/8.4 and issues TENSOR_LOAD_TO_LDS.
static __device__ __forceinline__ void tdm_load_2d_bf16(
    unsigned lds_addr, const void* gaddr,
    unsigned tensor_d0, unsigned tensor_d1, unsigned d0_stride,
    unsigned tile_d0, unsigned tile_d1) {
  unsigned long long ga = (unsigned long long)(uintptr_t)gaddr;
  v4u g0;
  g0[0] = 1u;                                    // count=1, no gather, user D#
  g0[1] = lds_addr;                              // LDS byte address
  g0[2] = (unsigned)(ga & 0xFFFFFFFFu);          // global_addr[31:0]
  g0[3] = (unsigned)((ga >> 32) & 0x1FFFFFFu)    // global_addr[56:32]
          | (2u << 30);                          // type=2 ("image")
  v8i g1;
  g1[0] = (int)(1u << 16);                       // wg_mask=0, data_size=1 (2B)
  g1[1] = (int)(tensor_d0 << 16);                // tensor_dim0[15:0] @63:48
  g1[2] = (int)((tensor_d0 >> 16) | (tensor_d1 << 16));   // dim0 hi | dim1 lo
  g1[3] = (int)((tensor_d1 >> 16) | (tile_d0 << 16));     // dim1 hi | tile_dim0
  g1[4] = (int)(tile_d1 & 0xFFFFu);              // tile_dim1; tile_dim2=0
  g1[5] = (int)d0_stride;                        // tensor_dim0_stride[31:0]
  g1[6] = 0;                                     // stride hi16 | dim1_stride lo
  g1[7] = 0;
  v4i g2 = {0, 0, 0, 0};
  v4i g3 = {0, 0, 0, 0};
#if defined(__clang_major__) && (__clang_major__ >= 23)
  v8i g4 = {0, 0, 0, 0, 0, 0, 0, 0};
  __builtin_amdgcn_tensor_load_to_lds(g0, g1, g2, g3, g4, 0);
#else
  __builtin_amdgcn_tensor_load_to_lds(g0, g1, g2, g3, 0);
#endif
}

// ---------- 1) fp32 -> bf16 conversion ----------
__global__ __launch_bounds__(256) void cvt_bf16_kernel(
    const float* __restrict__ x, unsigned short* __restrict__ y, int n4) {
  int i = blockIdx.x * blockDim.x + threadIdx.x;
  if (i < n4) {
    float4 f = ((const float4*)x)[i];
    uint2 o;
    o.x = f2bf_pk(f.x, f.y);
    o.y = f2bf_pk(f.z, f.w);
    ((uint2*)y)[i] = o;
  }
}

// ---------- 2) projection GEMM: Y = X @ W^T + b, stored per-head bf16 ----------
__global__ __launch_bounds__(256) void proj_bf16_kernel(
    const unsigned short* __restrict__ Xb,   // (MM, EE) bf16 row-major
    const unsigned short* __restrict__ Wb,   // (EE, EE) bf16 row-major
    const float* __restrict__ bias,          // (EE)
    unsigned short* __restrict__ Yb,         // bf16, (b,h,l,d) or (b,h,d,l)
    int transposed) {
  const int wid  = blockIdx.x * 8 + (threadIdx.x >> 5);
  const int lane = threadIdx.x & 31;
  const int lo = lane & 15, hi = lane >> 4;
  const int m0 = (wid >> 3) * 16;   // 256 m-tiles
  const int n0 = (wid & 7) * 64;    // 8 n-groups

  v8f acc[4];
#pragma unroll
  for (int c = 0; c < 4; c++) {
    float bv = bias[n0 + 16 * c + lo];
#pragma unroll
    for (int r = 0; r < 8; r++) acc[c][r] = bv;
  }

  for (int kk = 0; kk < EE; kk += 32) {
    v8u au;
#pragma unroll
    for (int v = 0; v < 8; v++) {
      int kb = 16 * (v >> 2) + 8 * hi + 2 * (v & 3);        // A-layout pair base
      au[v] = *(const unsigned*)&Xb[(m0 + lo) * EE + kk + kb];
    }
    v16bf A = asbf(au);
#pragma unroll
    for (int c = 0; c < 4; c++) {
      v8u bu;
      int n = n0 + 16 * c + lo;
#pragma unroll
      for (int v = 0; v < 8; v++) {
        int kb = 16 * (v >> 2) + 8 * hi + 2 * (v & 3);      // B-layout pair base
        bu[v] = *(const unsigned*)&Wb[n * EE + kk + kb];
      }
      acc[c] = wmma_bf16(A, asbf(bu), acc[c]);
    }
  }

#pragma unroll
  for (int c = 0; c < 4; c++) {
#pragma unroll
    for (int r = 0; r < 8; r++) {
      int mg = m0 + r + 8 * hi;
      int ng = n0 + 16 * c + lo;
      int b = mg >> 11, l = mg & (LL - 1);
      int h = ng >> 6,  d = ng & (HD - 1);
      long idx = transposed ? (((long)(b * HH + h) * HD + d) * LL + l)
                            : (((long)(b * HH + h) * LL + l) * HD + d);
      Yb[idx] = f2bf(acc[c][r]);
    }
  }
}

// ---------- 3) causal polynomial-kernel attention, TDM-fed streaming ----------
// One wave = 16 query rows of one (b,h). 32 keys per iteration; K/V chunks are
// DMA'd to LDS by the Tensor Data Mover, WMMA fragments read via ds_load.
__global__ __launch_bounds__(128) void attn_kernel(
    const unsigned short* __restrict__ Qb,   // (B,H,L,HD) bf16
    const unsigned short* __restrict__ Kb,   // (B,H,L,HD) bf16
    const unsigned short* __restrict__ Vt,   // (B,H,HD,L) bf16 (transposed)
    const float* __restrict__ log_tau,
    unsigned short* __restrict__ Ab) {       // (B*L, E) bf16
  __shared__ __align__(16) unsigned short Klds[4][32][HD];   // 16 KB: K chunk
  __shared__ __align__(16) unsigned short Vlds[4][HD][32];   // 16 KB: V chunk
  __shared__ __align__(16) unsigned short kvlds[4][16][34];  // padded kv tile

  const int wid  = blockIdx.x * 4 + (threadIdx.x >> 5);
  const int lane = threadIdx.x & 31;
  const int lo = lane & 15, hi = lane >> 4;
  const int slot = threadIdx.x >> 5;
  const int bh = wid >> 7;          // 0..15
  const int qt = wid & 127;         // query tile within L
  const int q0 = qt * 16;
  const int b  = bh >> 3, h = bh & 7;

  const unsigned short* Qr = Qb + (long)bh * LL * HD;
  const unsigned short* Kr = Kb + (long)bh * LL * HD;
  const unsigned short* Vr = Vt + (long)bh * HD * LL;

  // LDS byte addresses for the TDM descriptors (flat -> LDS truncates to [31:0])
  const unsigned k_lds = (unsigned)(uintptr_t)(void*)&Klds[slot][0][0];
  const unsigned v_lds = (unsigned)(uintptr_t)(void*)&Vlds[slot][0][0];

  const float tau = __expf(log_tau[0]);
  const float sc  = 1.0f / (8.0f * tau);     // 1/(sqrt(Hd)*tau)

  // Q fragments (K-dim = 64 -> two 32-chunks), loaded once from global
  v16bf aq[2];
#pragma unroll
  for (int ks = 0; ks < 2; ks++) {
    v8u u;
#pragma unroll
    for (int v = 0; v < 8; v++) {
      int kb = 16 * (v >> 2) + 8 * hi + 2 * (v & 3);
      u[v] = *(const unsigned*)&Qr[(q0 + lo) * HD + 32 * ks + kb];
    }
    aq[ks] = asbf(u);
  }

  v8u ou;
#pragma unroll
  for (int v = 0; v < 8; v++) ou[v] = 0x3F803F80u;   // bf16(1.0) pair
  const v16bf ONES = asbf(ou);

  v8f oacc[4];
#pragma unroll
  for (int c = 0; c < 4; c++) oacc[c] = vzero();
  v8f dacc = vzero();

  const int nj = (qt >> 1) + 1;   // 32-key chunks covering s <= q (causal)
  for (int j = 0; j < nj; j++) {
    const int s0 = 32 * j;

    // WAR: previous iteration's LDS reads must retire before TDM overwrites
    asm volatile("s_wait_dscnt 0" : : : "memory");

    // TDM: K tile (32 s-rows x 64 d) then V tile (64 d-rows x 32 s), in order
    tdm_load_2d_bf16(k_lds, &Kr[(long)s0 * HD], HD, LL, HD, HD, 32);
    tdm_load_2d_bf16(v_lds, &Vr[s0],            LL, HD, LL, 32, HD);

    if (s0 + 32 < LL) __builtin_prefetch(&Kr[(long)(s0 + 32) * HD], 0, 1);

    // K tile done when TENSORcnt <= 1 (same-wave TDM ops complete in order)
    __builtin_amdgcn_s_wait_tensorcnt(1);
    asm volatile("" : : : "memory");

    // ---- scores for two 16-key tiles, elementwise kernel, mask, stage kv
#pragma unroll
    for (int t = 0; t < 2; t++) {
      v8f p = vzero();
#pragma unroll
      for (int ks = 0; ks < 2; ks++) {
        v8u bu;
#pragma unroll
        for (int v = 0; v < 8; v++) {
          int kb = 16 * (v >> 2) + 8 * hi + 2 * (v & 3);
          bu[v] = *(const unsigned*)&Klds[slot][16 * t + lo][32 * ks + kb];
        }
        p = wmma_bf16(aq[ks], asbf(bu), p);
      }
      const int sg = s0 + 16 * t + lo;   // lane's key index (C/D: n = lane&15)
#pragma unroll
      for (int r = 0; r < 8; r++) {
        int mg = q0 + r + 8 * hi;                     // row (C/D layout)
        float x  = p[r] * sc;
        float qv = 1.0f / (1.0f + __expf(-x));        // sigmoid
        float kvv = fmaf(qv, fmaf(qv, fmaf(qv, 3.0f, -2.0f), 1.0f), 1.0f);
        kvv = fmaxf(kvv, 0.0f);
        kvv = (sg <= mg) ? kvv : 0.0f;                // causal mask
        kvlds[slot][r + 8 * hi][16 * t + lo] = f2bf(kvv);
      }
    }
    asm volatile("s_wait_dscnt 0" : : : "memory");    // cross-lane kv RAW

    // ---- re-read kv tile in A-fragment layout
    v8u ku;
#pragma unroll
    for (int v = 0; v < 8; v++) {
      int kb = 16 * (v >> 2) + 8 * hi + 2 * (v & 3);
      ku[v] = *(const unsigned*)&kvlds[slot][lo][kb];
    }
    v16bf KA = asbf(ku);

    // denom row-sums via WMMA against all-ones B (column 0 holds the sums)
    dacc = wmma_bf16(KA, ONES, dacc);

    // V tile fully landed in LDS
    __builtin_amdgcn_s_wait_tensorcnt(0);
    asm volatile("" : : : "memory");

    // out += kv @ V  (V tile is d-major in LDS -> k-contiguous B fragments)
#pragma unroll
    for (int c = 0; c < 4; c++) {
      v8u vu;
#pragma unroll
      for (int v = 0; v < 8; v++) {
        int kb = 16 * (v >> 2) + 8 * hi + 2 * (v & 3);
        vu[v] = *(const unsigned*)&Vlds[slot][16 * c + lo][kb];
      }
      oacc[c] = wmma_bf16(KA, asbf(vu), oacc[c]);
    }
    asm volatile("" : : : "memory");  // keep next iter after these LDS reads
  }

  // broadcast per-row denom (n==0 lives in lane 0 / lane 16 of each half)
  float invd[8];
#pragma unroll
  for (int r = 0; r < 8; r++) {
    float dm = __shfl(dacc[r], lane & 16, 32);
    invd[r] = 1.0f / fmaxf(dm, 1e-12f);
  }
#pragma unroll
  for (int c = 0; c < 4; c++) {
#pragma unroll
    for (int r = 0; r < 8; r++) {
      long row = (long)b * LL + q0 + r + 8 * hi;      // (B,L) row
      int  col = h * HD + 16 * c + lo;                // E column
      Ab[row * EE + col] = f2bf(oacc[c][r] * invd[r]);
    }
  }
}

// ---------- 4) output GEMM: d_out = Ab @ Wo^T + bo (fp32 out) ----------
__global__ __launch_bounds__(256) void out_gemm_kernel(
    const unsigned short* __restrict__ Ab,   // (MM, EE) bf16
    const unsigned short* __restrict__ Wb,   // Wo bf16 (EE, EE)
    const float* __restrict__ bias,
    float* __restrict__ Y) {                 // (MM, EE) fp32
  const int wid  = blockIdx.x * 8 + (threadIdx.x >> 5);
  const int lane = threadIdx.x & 31;
  const int lo = lane & 15, hi = lane >> 4;
  const int m0 = (wid >> 3) * 16;
  const int n0 = (wid & 7) * 64;

  v8f acc[4];
#pragma unroll
  for (int c = 0; c < 4; c++) {
    float bv = bias[n0 + 16 * c + lo];
#pragma unroll
    for (int r = 0; r < 8; r++) acc[c][r] = bv;
  }

  for (int kk = 0; kk < EE; kk += 32) {
    v8u au;
#pragma unroll
    for (int v = 0; v < 8; v++) {
      int kb = 16 * (v >> 2) + 8 * hi + 2 * (v & 3);
      au[v] = *(const unsigned*)&Ab[(m0 + lo) * EE + kk + kb];
    }
    v16bf A = asbf(au);
#pragma unroll
    for (int c = 0; c < 4; c++) {
      v8u bu;
      int n = n0 + 16 * c + lo;
#pragma unroll
      for (int v = 0; v < 8; v++) {
        int kb = 16 * (v >> 2) + 8 * hi + 2 * (v & 3);
        bu[v] = *(const unsigned*)&Wb[n * EE + kk + kb];
      }
      acc[c] = wmma_bf16(A, asbf(bu), acc[c]);
    }
  }

#pragma unroll
  for (int c = 0; c < 4; c++) {
#pragma unroll
    for (int r = 0; r < 8; r++) {
      long mg = m0 + r + 8 * hi;
      int  ng = n0 + 16 * c + lo;
      Y[mg * EE + ng] = acc[c][r];
    }
  }
}

// ---------- host ----------
extern "C" void kernel_launch(void* const* d_in, const int* in_sizes, int n_in,
                              void* d_out, int out_size, void* d_ws, size_t ws_size,
                              hipStream_t stream) {
  (void)in_sizes; (void)n_in; (void)out_size; (void)ws_size;

  const float* query   = (const float*)d_in[0];
  const float* key_    = (const float*)d_in[1];
  const float* value   = (const float*)d_in[2];
  const float* Wq      = (const float*)d_in[3];
  const float* bq      = (const float*)d_in[4];
  const float* Wk      = (const float*)d_in[5];
  const float* bk      = (const float*)d_in[6];
  const float* Wv      = (const float*)d_in[7];
  const float* bv      = (const float*)d_in[8];
  const float* Wo      = (const float*)d_in[9];
  const float* bo      = (const float*)d_in[10];
  const float* log_tau = (const float*)d_in[11];

  char* ws = (char*)d_ws;
  const size_t SZ_X = (size_t)MM * EE * 2;   // 4 MB  (bf16 activations)
  const size_t SZ_W = (size_t)EE * EE * 2;   // 512 KB (bf16 weights)
  unsigned short* Xqb = (unsigned short*)(ws + 0 * SZ_X);
  unsigned short* Xkb = (unsigned short*)(ws + 1 * SZ_X);
  unsigned short* Xvb = (unsigned short*)(ws + 2 * SZ_X);
  unsigned short* Wqb = (unsigned short*)(ws + 3 * SZ_X + 0 * SZ_W);
  unsigned short* Wkb = (unsigned short*)(ws + 3 * SZ_X + 1 * SZ_W);
  unsigned short* Wvb = (unsigned short*)(ws + 3 * SZ_X + 2 * SZ_W);
  unsigned short* Wob = (unsigned short*)(ws + 3 * SZ_X + 3 * SZ_W);
  char* p = ws + 3 * SZ_X + 4 * SZ_W;
  unsigned short* Qb = (unsigned short*)(p + 0 * SZ_X);  // (B,H,L,HD)
  unsigned short* Kb = (unsigned short*)(p + 1 * SZ_X);  // (B,H,L,HD)
  unsigned short* Vt = (unsigned short*)(p + 2 * SZ_X);  // (B,H,HD,L)
  unsigned short* Ab = (unsigned short*)(p + 3 * SZ_X);  // (B*L, E)

  const int nx4 = MM * EE / 4;   // 524288
  const int nw4 = EE * EE / 4;   // 65536
  cvt_bf16_kernel<<<(nx4 + 255) / 256, 256, 0, stream>>>(query, Xqb, nx4);
  cvt_bf16_kernel<<<(nx4 + 255) / 256, 256, 0, stream>>>(key_,  Xkb, nx4);
  cvt_bf16_kernel<<<(nx4 + 255) / 256, 256, 0, stream>>>(value, Xvb, nx4);
  cvt_bf16_kernel<<<(nw4 + 255) / 256, 256, 0, stream>>>(Wq, Wqb, nw4);
  cvt_bf16_kernel<<<(nw4 + 255) / 256, 256, 0, stream>>>(Wk, Wkb, nw4);
  cvt_bf16_kernel<<<(nw4 + 255) / 256, 256, 0, stream>>>(Wv, Wvb, nw4);
  cvt_bf16_kernel<<<(nw4 + 255) / 256, 256, 0, stream>>>(Wo, Wob, nw4);

  // 2048 waves each: 256 blocks x 256 threads (8 waves/block)
  proj_bf16_kernel<<<256, 256, 0, stream>>>(Xqb, Wqb, bq, Qb, 0);
  proj_bf16_kernel<<<256, 256, 0, stream>>>(Xkb, Wkb, bk, Kb, 0);
  proj_bf16_kernel<<<256, 256, 0, stream>>>(Xvb, Wvb, bv, Vt, 1);

  // 2048 waves: 512 blocks x 128 threads (4 waves/block)
  attn_kernel<<<512, 128, 0, stream>>>(Qb, Kb, Vt, log_tau, Ab);

  out_gemm_kernel<<<256, 256, 0, stream>>>(Ab, Wob, bo, (float*)d_out);
}
